// JointNet_53085795778975
// MI455X (gfx1250) — matrix operand order repaired and checked by reading, not compiled, not run on hardware
//
#include <hip/hip_runtime.h>
#include <hip/hip_bf16.h>

typedef __attribute__((ext_vector_type(16))) __bf16 v16bf;
typedef __attribute__((ext_vector_type(8)))  __bf16 v8bf;
typedef __attribute__((ext_vector_type(4)))  __bf16 v4bf;
typedef __attribute__((ext_vector_type(8)))  float  v8f;

#define B_     8
#define T_     200
#define U_     50
#define INNER_ 640
#define VOCAB_ 4096
#define MROWS  (B_ * T_ * U_)   // 80000
#define KSTEPS (INNER_ / 32)    // 20

#if defined(__has_builtin)
# if __has_builtin(__builtin_amdgcn_tanhf)
#  define FAST_TANH(x) __builtin_amdgcn_tanhf(x)
# else
#  define FAST_TANH(x) tanhf(x)
# endif
#else
# define FAST_TANH(x) tanhf(x)
#endif

// ---------------------------------------------------------------------------
// Kernel 1: convert W_proj (fp32, [VOCAB][INNER]) -> bf16, once.
// ---------------------------------------------------------------------------
__global__ __launch_bounds__(256) void cvt_wproj(const float* __restrict__ W,
                                                 __bf16* __restrict__ Wb) {
    const int gid = blockIdx.x * 256 + threadIdx.x;     // exactly VOCAB*INNER/4 threads
    const float4 f = ((const float4*)W)[gid];
    v4bf o;
    o[0] = (__bf16)f.x; o[1] = (__bf16)f.y;
    o[2] = (__bf16)f.z; o[3] = (__bf16)f.w;
    *(v4bf*)(Wb + (size_t)gid * 4) = o;
}

// ---------------------------------------------------------------------------
// Kernel 2: Y[r][i] = sum_k X[r][k] * W[i][k] + bias[i]
// 16 rows per block (amortizes W traffic through L2), 256 threads.
// Used for enc_p (1600 rows) and dec_p (400 rows); both multiples of 16.
// ---------------------------------------------------------------------------
__global__ __launch_bounds__(256) void proj16(const float* __restrict__ X,
                                              const float* __restrict__ W,
                                              const float* __restrict__ bias,
                                              float* __restrict__ Y) {
    __shared__ float4 xs4[16 * (INNER_ / 4)];           // 16 rows * 160 float4 = 40 KB
    const int tid = threadIdx.x;
    const size_t rowBase = (size_t)blockIdx.x * 16;
    const float4* X4 = (const float4*)(X + rowBase * INNER_);
#pragma unroll
    for (int i = 0; i < 10; ++i) xs4[tid + i * 256] = X4[tid + i * 256];
    __syncthreads();

    for (int i = tid; i < INNER_; i += 256) {
        const float4* Wr = (const float4*)(W + (size_t)i * INNER_);
        float acc[16];
#pragma unroll
        for (int r = 0; r < 16; ++r) acc[r] = 0.f;
        for (int k4 = 0; k4 < INNER_ / 4; ++k4) {
            const float4 w = Wr[k4];
#pragma unroll
            for (int r = 0; r < 16; ++r) {
                const float4 x = xs4[r * (INNER_ / 4) + k4];
                acc[r] += x.x * w.x + x.y * w.y + x.z * w.z + x.w * w.w;
            }
        }
        const float bi = bias[i];
#pragma unroll
        for (int r = 0; r < 16; ++r)
            Y[(rowBase + r) * INNER_ + i] = acc[r] + bi;
    }
}

// ---------------------------------------------------------------------------
// Kernel 3: fused  out[m][n] = sum_k tanh(enc_p[bt][k]*dec_p[bu][k]) * Wp[n][k]
// WG tile: 64 (M) x 512 (N); 8 waves, each wave 64x64 via 4x4 WMMA 16x16x32 bf16.
// A tile (64x32 bf16) built cooperatively each K-step (tanh fused) into a
// DOUBLE-BUFFERED LDS tile -> one barrier per K-step, A-gen overlaps WMMA.
// B fragments loaded per-lane straight from bf16 W_proj (L2-resident).
// Output written with non-temporal stores (streaming, never re-read).
// ---------------------------------------------------------------------------
__global__ __launch_bounds__(256) void joint_gemm(const float*  __restrict__ encP,
                                                  const float*  __restrict__ decP,
                                                  const __bf16* __restrict__ Wb,
                                                  float*        __restrict__ out) {
    constexpr int AS   = 40;                   // LDS row stride in bf16 elems (80 B)
    constexpr int ABUF = 64 * AS;              // one tile = 2560 elems (5120 B)
    __shared__ __bf16 Asm[2 * ABUF];           // ping-pong, 10 KB

    const int tid  = threadIdx.x;
    const int wave = tid >> 5;
    const int lane = tid & 31;
    const int mBase = blockIdx.y * 64;
    const int nBase = blockIdx.x * 512 + wave * 64;

    // --- A-generation mapping: this thread produces A[aRow][aK..aK+7] each step
    const int aRow = tid >> 2;                 // 0..63
    const int aK   = (tid & 3) * 8;            // 0,8,16,24
    const int gr   = mBase + aRow;             // global M row
    const int bt   = gr / U_;                  // b*T + t
    const int u    = gr - bt * U_;
    const int b    = bt / T_;
    const float* encRow = encP + (size_t)bt * INNER_;
    const float* decRow = decP + ((size_t)b * U_ + u) * INNER_;
    __bf16* aDst = &Asm[aRow * AS + aK];

    // --- fragment-load mapping (ISA 7.12.2 wave32 layouts)
    const int frow = lane & 15;                // A row / B column within tile
    const int klo  = (lane >> 4) * 8;          // A: K-phase 0 or 8
    const int bklo = (lane >> 4) * 8;          // B: K-phase 0 or 8
    const __bf16* aSrc0 = &Asm[frow * AS + klo];

    v8f acc[4][4];
#pragma unroll
    for (int mt = 0; mt < 4; ++mt)
#pragma unroll
        for (int nt = 0; nt < 4; ++nt)
#pragma unroll
            for (int j = 0; j < 8; ++j) acc[mt][nt][j] = 0.f;

    // ---- prologue: build tile 0 into buffer 0
    {
        const float4 e0 = *(const float4*)(encRow + aK);
        const float4 e1 = *(const float4*)(encRow + aK + 4);
        const float4 d0 = *(const float4*)(decRow + aK);
        const float4 d1 = *(const float4*)(decRow + aK + 4);
        v8bf pk;
        pk[0] = (__bf16)FAST_TANH(e0.x * d0.x);
        pk[1] = (__bf16)FAST_TANH(e0.y * d0.y);
        pk[2] = (__bf16)FAST_TANH(e0.z * d0.z);
        pk[3] = (__bf16)FAST_TANH(e0.w * d0.w);
        pk[4] = (__bf16)FAST_TANH(e1.x * d1.x);
        pk[5] = (__bf16)FAST_TANH(e1.y * d1.y);
        pk[6] = (__bf16)FAST_TANH(e1.z * d1.z);
        pk[7] = (__bf16)FAST_TANH(e1.w * d1.w);
        *(v8bf*)aDst = pk;
    }
    __syncthreads();

    for (int it = 0; it < KSTEPS; ++it) {
        const int k0   = it * 32;
        const int cur  = (it & 1) * ABUF;
        const int nxt  = ((it + 1) & 1) * ABUF;

        // B fragments from global bf16 W_proj: K = bklo..+7 and 16+bklo..+7 of column n
        union { v16bf v; v8bf h[2]; } bfm[4];
#pragma unroll
        for (int nt = 0; nt < 4; ++nt) {
            const __bf16* p = Wb + (size_t)(nBase + nt * 16 + frow) * INNER_ + k0 + bklo;
            bfm[nt].h[0] = *(const v8bf*)p;
            bfm[nt].h[1] = *(const v8bf*)(p + 16);
        }
        // A fragments from LDS buffer `cur`
        union { v16bf v; v8bf h[2]; } af[4];
#pragma unroll
        for (int mt = 0; mt < 4; ++mt) {
            const __bf16* p = aSrc0 + cur + mt * 16 * AS;
            af[mt].h[0] = *(const v8bf*)p;
            af[mt].h[1] = *(const v8bf*)(p + 16);
        }

        // produce NEXT A tile into the other buffer (overlaps with WMMAs below)
        if (it + 1 < KSTEPS) {
            const int kn = k0 + 32;
            const float4 e0 = *(const float4*)(encRow + kn + aK);
            const float4 e1 = *(const float4*)(encRow + kn + aK + 4);
            const float4 d0 = *(const float4*)(decRow + kn + aK);
            const float4 d1 = *(const float4*)(decRow + kn + aK + 4);
            v8bf pk;
            pk[0] = (__bf16)FAST_TANH(e0.x * d0.x);
            pk[1] = (__bf16)FAST_TANH(e0.y * d0.y);
            pk[2] = (__bf16)FAST_TANH(e0.z * d0.z);
            pk[3] = (__bf16)FAST_TANH(e0.w * d0.w);
            pk[4] = (__bf16)FAST_TANH(e1.x * d1.x);
            pk[5] = (__bf16)FAST_TANH(e1.y * d1.y);
            pk[6] = (__bf16)FAST_TANH(e1.z * d1.z);
            pk[7] = (__bf16)FAST_TANH(e1.w * d1.w);
            *(v8bf*)(aDst + nxt) = pk;
        }

#pragma unroll
        for (int mt = 0; mt < 4; ++mt)
#pragma unroll
            for (int nt = 0; nt < 4; ++nt)
                acc[mt][nt] = __builtin_amdgcn_wmma_f32_16x16x32_bf16(
                    false, af[mt].v, false, bfm[nt].v,
                    (short)0, acc[mt][nt], false, false);

        // single barrier: RAW on next tile + WAR on current tile
        __syncthreads();
    }

    // C store: VGPR j -> M = j + (lane>=16)*8, N = lane%16  (non-temporal)
    const int mOff = (lane >> 4) * 8;
    const int nCol = lane & 15;
#pragma unroll
    for (int mt = 0; mt < 4; ++mt) {
#pragma unroll
        for (int nt = 0; nt < 4; ++nt) {
            const size_t n = (size_t)(nBase + nt * 16 + nCol);
#pragma unroll
            for (int j = 0; j < 8; ++j) {
                const size_t m = (size_t)(mBase + mt * 16 + mOff + j);
                __builtin_nontemporal_store(acc[mt][nt][j], &out[m * VOCAB_ + n]);
            }
        }
    }
}

// ---------------------------------------------------------------------------
extern "C" void kernel_launch(void* const* d_in, const int* in_sizes, int n_in,
                              void* d_out, int out_size, void* d_ws, size_t ws_size,
                              hipStream_t stream) {
    const float* enc_state = (const float*)d_in[0];   // [B,T,ENC]
    const float* dec_state = (const float*)d_in[1];   // [B,U,DEC]
    const float* W_enc     = (const float*)d_in[2];   // [INNER,ENC]
    const float* b_enc     = (const float*)d_in[3];   // [INNER]
    const float* W_dec     = (const float*)d_in[4];   // [INNER,DEC]
    const float* b_dec     = (const float*)d_in[5];   // [INNER]
    const float* W_proj    = (const float*)d_in[6];   // [VOCAB,INNER]
    float* out = (float*)d_out;

    // workspace layout: enc_p fp32 | dec_p fp32 | W_proj bf16   (~10.4 MB)
    float*  encP = (float*)d_ws;
    float*  decP = encP + (size_t)B_ * T_ * INNER_;
    __bf16* Wb   = (__bf16*)(decP + (size_t)B_ * U_ * INNER_);

    cvt_wproj<<<(VOCAB_ * INNER_ / 4) / 256, 256, 0, stream>>>(W_proj, Wb);
    proj16<<<(B_ * T_) / 16, 256, 0, stream>>>(enc_state, W_enc, b_enc, encP);
    proj16<<<(B_ * U_) / 16, 256, 0, stream>>>(dec_state, W_dec, b_dec, decP);

    dim3 grid(VOCAB_ / 512, MROWS / 64);   // (8, 1250)
    joint_gemm<<<grid, 256, 0, stream>>>(encP, decP, Wb, out);
}
